// JointNet_20684562497747
// MI455X (gfx1250) — compile-verified, matrix operand order checked by reading, not run on hardware
//
#include <hip/hip_runtime.h>
#include <math.h>

typedef _Float16 v4h  __attribute__((ext_vector_type(4)));
typedef _Float16 v8h  __attribute__((ext_vector_type(8)));
typedef _Float16 v16h __attribute__((ext_vector_type(16)));
typedef float    v8f  __attribute__((ext_vector_type(8)));

#define B_      8
#define T_      200
#define U_      50
#define D_      512
#define INNER_  640
#define VOCAB_  1024
#define M_TOTAL (B_ * T_ * U_)   /* 80000 */
#define TM      64               /* rows per workgroup: 80000/64 = 1250 WGs */
#define XS      (D_ + 8)         /* 520 halves: 1040B row stride -> 4-bank rotate */
#define HS      (INNER_ + 8)     /* 648 halves: 1296B row stride -> 4-bank rotate */

// gfx1250 hardware tanh (V_TANH_F32, TRANS pipe — co-executes with WMMA).
__device__ __forceinline__ float fast_tanh(float x) {
#if __has_builtin(__builtin_amdgcn_tanhf)
    return __builtin_amdgcn_tanhf(x);
#elif __has_builtin(__builtin_amdgcn_tanh_f32)
    return __builtin_amdgcn_tanh_f32(x);
#else
    return tanhf(x);
#endif
}

// ---------------- weight pre-pass: f32 -> f16, transposed ----------------
// W1 (D x INNER) -> W1t (INNER x D), so B-fragment loads use lane = N column.
__global__ void cvt_w1(const float* __restrict__ W1, _Float16* __restrict__ W1t) {
    int idx = blockIdx.x * 256 + threadIdx.x;
    if (idx >= D_ * INNER_) return;
    int d = idx / INNER_, h = idx % INNER_;
    W1t[h * D_ + d] = (_Float16)W1[idx];
}
// W2 (INNER x VOCAB) -> W2t (VOCAB x INNER)
__global__ void cvt_w2(const float* __restrict__ W2, _Float16* __restrict__ W2t) {
    int idx = blockIdx.x * 256 + threadIdx.x;
    if (idx >= INNER_ * VOCAB_) return;
    int i = idx / VOCAB_, v = idx % VOCAB_;
    W2t[v * INNER_ + i] = (_Float16)W2[idx];
}

// lane<16: halves 0..7 = K kb+0..7,  halves 8..15 = K kb+16..23
// lane>=16: halves 0..7 = K kb+8..15, halves 8..15 = K kb+24..31
__device__ __forceinline__ v16h load_frag(const _Float16* p) {
    v8h lo = *(const v8h*)(p);
    v8h hi = *(const v8h*)(p + 16);
    return __builtin_shufflevector(lo, hi, 0,1,2,3,4,5,6,7,8,9,10,11,12,13,14,15);
}

// ---------------- fused joint-network kernel ----------------
__launch_bounds__(256)
__global__ void joint_fused(const float* __restrict__ enc,
                            const float* __restrict__ dec,
                            const float* __restrict__ b1,
                            const float* __restrict__ b2,
                            const _Float16* __restrict__ W1t,
                            const _Float16* __restrict__ W2t,
                            float* __restrict__ out) {
    __shared__ _Float16 Xs[TM * XS];   // 66,560 B
    __shared__ _Float16 Hs[TM * HS];   // 82,944 B

    const int tid   = threadIdx.x;
    const int wave  = tid >> 5;
    const int lane  = tid & 31;
    const int rlane = lane & 15;       // N/M position within half-wave
    const int hsel  = lane >> 4;       // which K-half this lane holds
    const int Rbase = blockIdx.x * TM;

    // ---- build X = enc_row + dec_row, f32 -> f16, into LDS ----
    for (int i = tid; i < TM * (D_ / 4); i += 256) {
        int row = i / (D_ / 4);
        int k4  = (i % (D_ / 4)) * 4;
        int R   = Rbase + row;
        int b   = R / (T_ * U_);
        int t   = (R / U_) % T_;
        int u   = R % U_;
        const float4 e  = *(const float4*)(enc + ((size_t)(b * T_ + t)) * D_ + k4);
        const float4 dv = *(const float4*)(dec + ((size_t)(b * U_ + u)) * D_ + k4);
        v4h x;
        x[0] = (_Float16)(e.x + dv.x);
        x[1] = (_Float16)(e.y + dv.y);
        x[2] = (_Float16)(e.z + dv.z);
        x[3] = (_Float16)(e.w + dv.w);
        *(v4h*)&Xs[row * XS + k4] = x;
    }
    __syncthreads();

    // ---- GEMM1: hidden = tanh(X @ W1 + b1), result f16 in LDS ----
    for (int nt = wave; nt < INNER_ / 16; nt += 8) {
        const int   ncol = nt * 16 + rlane;
        const float bias = b1[ncol];
        const _Float16* Bp = W1t + (size_t)ncol * D_;
        v8f acc0 = {}, acc1 = {}, acc2 = {}, acc3 = {};
        for (int kb = 0; kb < D_; kb += 32) {
            v16h bf = load_frag(Bp + kb + hsel * 8);
            v16h a0 = load_frag(&Xs[( 0 + rlane) * XS + kb + hsel * 8]);
            v16h a1 = load_frag(&Xs[(16 + rlane) * XS + kb + hsel * 8]);
            v16h a2 = load_frag(&Xs[(32 + rlane) * XS + kb + hsel * 8]);
            v16h a3 = load_frag(&Xs[(48 + rlane) * XS + kb + hsel * 8]);
            acc0 = __builtin_amdgcn_wmma_f32_16x16x32_f16(false, a0, false, bf, (short)0, acc0, false, false);
            acc1 = __builtin_amdgcn_wmma_f32_16x16x32_f16(false, a1, false, bf, (short)0, acc1, false, false);
            acc2 = __builtin_amdgcn_wmma_f32_16x16x32_f16(false, a2, false, bf, (short)0, acc2, false, false);
            acc3 = __builtin_amdgcn_wmma_f32_16x16x32_f16(false, a3, false, bf, (short)0, acc3, false, false);
        }
        // C layout: vgpr j -> M = j + 8*hsel, N = rlane
        #pragma unroll
        for (int j = 0; j < 8; ++j) {
            int mrow = j + hsel * 8;
            Hs[( 0 + mrow) * HS + ncol] = (_Float16)fast_tanh(acc0[j] + bias);
            Hs[(16 + mrow) * HS + ncol] = (_Float16)fast_tanh(acc1[j] + bias);
            Hs[(32 + mrow) * HS + ncol] = (_Float16)fast_tanh(acc2[j] + bias);
            Hs[(48 + mrow) * HS + ncol] = (_Float16)fast_tanh(acc3[j] + bias);
        }
    }
    __syncthreads();

    // ---- GEMM2: logits = hidden @ W2 + b2, f32 to global ----
    for (int nt = wave; nt < VOCAB_ / 16; nt += 8) {
        const int   ncol = nt * 16 + rlane;
        const float bias = b2[ncol];
        const _Float16* Bp = W2t + (size_t)ncol * INNER_;
        v8f acc0 = {}, acc1 = {}, acc2 = {}, acc3 = {};
        for (int kb = 0; kb < INNER_; kb += 32) {
            v16h bf = load_frag(Bp + kb + hsel * 8);
            v16h a0 = load_frag(&Hs[( 0 + rlane) * HS + kb + hsel * 8]);
            v16h a1 = load_frag(&Hs[(16 + rlane) * HS + kb + hsel * 8]);
            v16h a2 = load_frag(&Hs[(32 + rlane) * HS + kb + hsel * 8]);
            v16h a3 = load_frag(&Hs[(48 + rlane) * HS + kb + hsel * 8]);
            acc0 = __builtin_amdgcn_wmma_f32_16x16x32_f16(false, a0, false, bf, (short)0, acc0, false, false);
            acc1 = __builtin_amdgcn_wmma_f32_16x16x32_f16(false, a1, false, bf, (short)0, acc1, false, false);
            acc2 = __builtin_amdgcn_wmma_f32_16x16x32_f16(false, a2, false, bf, (short)0, acc2, false, false);
            acc3 = __builtin_amdgcn_wmma_f32_16x16x32_f16(false, a3, false, bf, (short)0, acc3, false, false);
        }
        #pragma unroll
        for (int j = 0; j < 8; ++j) {
            int mrow = j + hsel * 8;
            out[((size_t)(Rbase +  0 + mrow)) * VOCAB_ + ncol] = acc0[j] + bias;
            out[((size_t)(Rbase + 16 + mrow)) * VOCAB_ + ncol] = acc1[j] + bias;
            out[((size_t)(Rbase + 32 + mrow)) * VOCAB_ + ncol] = acc2[j] + bias;
            out[((size_t)(Rbase + 48 + mrow)) * VOCAB_ + ncol] = acc3[j] + bias;
        }
    }
}

extern "C" void kernel_launch(void* const* d_in, const int* in_sizes, int n_in,
                              void* d_out, int out_size, void* d_ws, size_t ws_size,
                              hipStream_t stream) {
    const float* enc = (const float*)d_in[0];
    const float* dec = (const float*)d_in[1];
    const float* W1  = (const float*)d_in[2];
    const float* b1  = (const float*)d_in[3];
    const float* W2  = (const float*)d_in[4];
    const float* b2  = (const float*)d_in[5];

    _Float16* W1t = (_Float16*)d_ws;                                    // 640*512*2  = 655,360 B
    _Float16* W2t = (_Float16*)((char*)d_ws + (size_t)D_ * INNER_ * 2); // 1024*640*2 = 1,310,720 B

    cvt_w1<<<(D_ * INNER_ + 255) / 256, 256, 0, stream>>>(W1, W1t);
    cvt_w2<<<(INNER_ * VOCAB_ + 255) / 256, 256, 0, stream>>>(W2, W2t);
    joint_fused<<<M_TOTAL / TM, 256, 0, stream>>>(enc, dec, b1, b2, W1t, W2t, (float*)d_out);
}
